// RefractionSheet_77790447665460
// MI455X (gfx1250) — matrix operand order repaired and checked by reading, not compiled, not run on hardware
//
#include <hip/hip_runtime.h>

typedef _Float16 f16;
typedef __attribute__((ext_vector_type(16))) _Float16 v16h;
typedef __attribute__((ext_vector_type(8)))  _Float16 v8h;
typedef __attribute__((ext_vector_type(8)))  float    v8f;

#define ACT_NONE      0
#define ACT_RELU      1
#define ACT_SIN       2
#define ACT_SIN_RELU  3
#define ACT_TANH_CLIP 4
#define ACT_SIGMOID   5

#define CLAMP_V   25.133f
#define MAX_ANG   0.0349065850398866f   /* radians(2.0) */
#define TBL_SIZE  524288u
#define PRIME_Y   2654435761u

__constant__ int LRES[16] = {32,44,61,84,116,161,222,307,425,588,
                             812,1123,1551,2144,2963,4095};

__device__ __forceinline__ v16h cat16(v8h a, v8h b) {
  return __builtin_shufflevector(a, b, 0,1,2,3,4,5,6,7,8,9,10,11,12,13,14,15);
}

__device__ __forceinline__ float act_apply(int ACT, float v) {
  if (ACT == ACT_RELU)           return fmaxf(v, 0.f);
  else if (ACT == ACT_SIN)       return __sinf(fminf(fmaxf(v, -CLAMP_V), CLAMP_V));
  else if (ACT == ACT_SIN_RELU)  return fmaxf(__sinf(fminf(fmaxf(v, -CLAMP_V), CLAMP_V)), 0.f);
  else if (ACT == ACT_TANH_CLIP) return fminf(fmaxf(tanhf(v), -1.f), 1.f);
  else if (ACT == ACT_SIGMOID)   return 1.f / (1.f + __expf(-v));
  return v;
}

// ---------------------------------------------------------------------------
// Fused WMMA GEMM:  Y[M x Npad] = act( X[M x Kpad] * Wh^T + bias )
// Wh row-major [Npad][Kpad] f16 (zero padded), bias f32 [Npad].
// Block = 256 threads (8 waves). Block tile = 32 rows x 256 cols.
// A tile (32 x Kpad) staged in LDS once per block (<=48KB), read via
// ds_load_b128; each wave computes a 32x32 tile = 4 WMMA accumulators,
// so each K-step issues 4 ds_load_b128 + 4 global b128 for 4 WMMAs.
// Fragment layouts per CDNA5 ISA 7.12.2 (wave32):
//   A 16x32 f16 : lane L: M=L&15; halfs 0..7 -> K=kk+(L>=16?8:0)+j,
//                                halfs 8..15 -> K=kk+16+(L>=16?8:0)+j
//   B 32x16 f16 : lane L: N=L&15; same K mapping as A
//   C/D 16x16 f32: lane L: N=L&15; vgpr r   -> M=r+(L>=16?8:0)
// ---------------------------------------------------------------------------
template<int ACT, bool F16OUT>
__global__ __launch_bounds__(256) void gemm_wmma(
    const f16* __restrict__ X, int ldx,
    const f16* __restrict__ Wh, const float* __restrict__ bias,
    void* __restrict__ Yv, int ldy, int Nvalid, int Kpad, int Npad)
{
  extern __shared__ f16 lds[];
  const int tid  = threadIdx.x;
  const int lane = tid & 31;
  const int wave = tid >> 5;
  const int hi   = lane >> 4;
  const int lm   = lane & 15;
  const int rowbase = blockIdx.x << 5;                 // 32 rows / block
  const int ncol0   = (blockIdx.y << 8) + (wave << 5); // 32 cols / wave

  // ---- cooperative stage: X[rowbase..rowbase+31][0..Kpad) -> LDS ----
  const int KV = Kpad >> 3;          // v8h per row
  const int TV = KV << 5;            // total v8h (32 rows)
  for (int i = tid; i < TV; i += 256) {
    int rr = i / KV, cc = i - rr * KV;
    ((v8h*)lds)[i] = *(const v8h*)(X + (size_t)(rowbase + rr) * ldx + (cc << 3));
  }
  __syncthreads();

  if (ncol0 >= Npad) return;
  const bool two = (ncol0 + 16) < Npad;

  const f16* bp0 = Wh + (size_t)(ncol0 + lm) * Kpad + hi * 8;
  const f16* bp1 = two ? (bp0 + (size_t)16 * Kpad) : bp0;  // alias when N-tile 1 unused
  const f16* la0 = lds + (size_t)lm * Kpad + hi * 8;
  const f16* la1 = la0 + (size_t)16 * Kpad;

  v8f acc00 = {}, acc01 = {}, acc10 = {}, acc11 = {};
  for (int kk = 0; kk < Kpad; kk += 32) {
    v16h a0 = cat16(*(const v8h*)(la0 + kk), *(const v8h*)(la0 + kk + 16));
    v16h a1 = cat16(*(const v8h*)(la1 + kk), *(const v8h*)(la1 + kk + 16));
    v16h b0 = cat16(*(const v8h*)(bp0 + kk), *(const v8h*)(bp0 + kk + 16));
    v16h b1 = cat16(*(const v8h*)(bp1 + kk), *(const v8h*)(bp1 + kk + 16));
    acc00 = __builtin_amdgcn_wmma_f32_16x16x32_f16(false, a0, false, b0, (short)0, acc00, false, false);
    acc01 = __builtin_amdgcn_wmma_f32_16x16x32_f16(false, a0, false, b1, (short)0, acc01, false, false);
    acc10 = __builtin_amdgcn_wmma_f32_16x16x32_f16(false, a1, false, b0, (short)0, acc10, false, false);
    acc11 = __builtin_amdgcn_wmma_f32_16x16x32_f16(false, a1, false, b1, (short)0, acc11, false, false);
  }

  float* yf = (float*)Yv;
  f16*   yh = (f16*)Yv;
#pragma unroll
  for (int nt = 0; nt < 2; ++nt) {
    const int n = ncol0 + (nt << 4) + lm;
    if (n >= Npad) break;
    const float bb = bias[n];
#pragma unroll
    for (int mt = 0; mt < 2; ++mt) {
      const v8f acc = (nt == 0) ? (mt == 0 ? acc00 : acc10)
                                : (mt == 0 ? acc01 : acc11);
#pragma unroll
      for (int r = 0; r < 8; ++r) {
        float v = act_apply(ACT, acc[r] + bb);
        const int row = rowbase + (mt << 4) + r + (hi << 3);
        if (F16OUT)          yh[(size_t)row * ldy + n] = (f16)v;
        else if (n < Nvalid) yf[(size_t)row * ldy + n] = v;
      }
    }
  }
}

// ---------------------------------------------------------------------------
// Weight / bias packing: f32 (N,Kin) -> f16 (Npad,Kpad) zero padded
// ---------------------------------------------------------------------------
__global__ void pack_weight(const float* __restrict__ W, f16* __restrict__ Wh,
                            int Kin, int Kpad, int N, int Npad)
{
  int idx = blockIdx.x * 256 + threadIdx.x;
  if (idx >= Npad * Kpad) return;
  int n = idx / Kpad, k = idx - n * Kpad;
  float v = (n < N && k < Kin) ? W[(size_t)n * Kin + k] : 0.f;
  Wh[idx] = (f16)v;
}

__global__ void pack_bias(const float* __restrict__ b, float* __restrict__ bp,
                          int N, int Npad)
{
  int n = blockIdx.x * 256 + threadIdx.x;
  if (n >= Npad) return;
  bp[n] = (n < N) ? b[n] : 0.f;
}

// ---------------------------------------------------------------------------
// Hashgrid encode + cam_feat concat -> h0 (chunk x 64 f16)
// ---------------------------------------------------------------------------
__global__ void hash_concat(const float* __restrict__ xs, const float* __restrict__ ys,
                            const float* __restrict__ cam, const float* __restrict__ table,
                            f16* __restrict__ h0, int rayBase, int nRays)
{
  int t = blockIdx.x * 256 + threadIdx.x;
  if (t >= nRays) return;
  int g = rayBase + t;
  float px = fminf(fmaxf(rintf(xs[g]), 0.f), 511.f);
  float py = fminf(fmaxf(rintf(ys[g]), 0.f), 511.f);
  float ux = px * (1.f / 512.f), uy = py * (1.f / 512.f);
  f16* out = h0 + (size_t)t * 64;
#pragma unroll
  for (int l = 0; l < 16; ++l) {
    int res = LRES[l];
    float fx = ux * res, fy = uy * res;
    float x0f = floorf(fx), y0f = floorf(fy);
    int x0 = (int)x0f, y0 = (int)y0f;
    float wx = fx - x0f, wy = fy - y0f;
    bool dense = ((res + 1) * (res + 1) <= (int)TBL_SIZE);
    float f0 = 0.f, f1 = 0.f;
#pragma unroll
    for (int dy = 0; dy < 2; ++dy)
#pragma unroll
      for (int dx = 0; dx < 2; ++dx) {
        int cx = min(max(x0 + dx, 0), res);
        int cy = min(max(y0 + dy, 0), res);
        unsigned idx = dense ? (unsigned)(cx + cy * (res + 1))
                             : ((((unsigned)cx) ^ ((unsigned)cy * PRIME_Y)) & (TBL_SIZE - 1u));
        const float* e = table + ((size_t)l * TBL_SIZE + idx) * 2;
        float w = (dx ? wx : 1.f - wx) * (dy ? wy : 1.f - wy);
        f0 += w * e[0];
        f1 += w * e[1];
      }
    out[2 * l]     = (f16)f0;
    out[2 * l + 1] = (f16)f1;
  }
  const float* c = cam + (size_t)g * 32;
#pragma unroll
  for (int j = 0; j < 32; ++j) out[32 + j] = (f16)c[j];
}

// ---------------------------------------------------------------------------
// Build inp (chunk x 3 x 512): [base(384) | pe_dir(51) | pe_loc(51) | 0(26)]
// One thread per (ray, slot). Reads delta_raw (already in d_out).
// ---------------------------------------------------------------------------
__global__ void build_inp(const float* __restrict__ delta_all,
                          const float* __restrict__ rd, const float* __restrict__ ro,
                          const f16* __restrict__ baseb, f16* __restrict__ inp,
                          int rayBase, int nRays)
{
  int t = blockIdx.x * 256 + threadIdx.x;
  if (t >= nRays * 3) return;
  int r = t / 3, s = t - r * 3;
  int g = rayBase + r;
  f16* ip = inp + (size_t)(r * 3 + s) * 512;

  const v8h* src = (const v8h*)(baseb + (size_t)r * 384);
  v8h* dst = (v8h*)ip;
#pragma unroll 8
  for (int j = 0; j < 48; ++j) dst[j] = src[j];

  const float* del = delta_all + (size_t)g * 9 + s * 3;
  float dx = rd[g * 3 + 0] + MAX_ANG * del[0];
  float dy = rd[g * 3 + 1] + MAX_ANG * del[1];
  float dz = rd[g * 3 + 2] + MAX_ANG * del[2];
  float nrm = fmaxf(sqrtf(dx * dx + dy * dy + dz * dz), 1e-12f);
  float inv = 1.f / nrm;
  dx *= inv; dy *= inv; dz *= inv;

  ip[384] = (f16)dx; ip[385] = (f16)dy; ip[386] = (f16)dz;
  float k = 1.f;
#pragma unroll
  for (int b = 0; b < 8; ++b) {
    ip[387 + b * 6 + 0] = (f16)__sinf(k * dx);
    ip[387 + b * 6 + 1] = (f16)__sinf(k * dy);
    ip[387 + b * 6 + 2] = (f16)__sinf(k * dz);
    ip[387 + b * 6 + 3] = (f16)__cosf(k * dx);
    ip[387 + b * 6 + 4] = (f16)__cosf(k * dy);
    ip[387 + b * 6 + 5] = (f16)__cosf(k * dz);
    k *= 2.f;
  }
  float ox = ro[g * 3 + 0], oy = ro[g * 3 + 1], oz = ro[g * 3 + 2];
  ip[435] = (f16)ox; ip[436] = (f16)oy; ip[437] = (f16)oz;
  k = 1.f;
#pragma unroll
  for (int b = 0; b < 8; ++b) {
    ip[438 + b * 6 + 0] = (f16)__sinf(k * ox);
    ip[438 + b * 6 + 1] = (f16)__sinf(k * oy);
    ip[438 + b * 6 + 2] = (f16)__sinf(k * oz);
    ip[438 + b * 6 + 3] = (f16)__cosf(k * ox);
    ip[438 + b * 6 + 4] = (f16)__cosf(k * oy);
    ip[438 + b * 6 + 5] = (f16)__cosf(k * oz);
    k *= 2.f;
  }
#pragma unroll
  for (int j = 486; j < 512; ++j) ip[j] = (f16)0.f;
}

// ---------------------------------------------------------------------------
// Channel mix: col[b,o,k] = sum_i mix[o,i] * sigmoid_col[b,i,k]
// ---------------------------------------------------------------------------
__global__ void mix_sig(const float* __restrict__ colbuf, const float* __restrict__ mix,
                        float* __restrict__ outcol, int rayBase, int nRays)
{
  int t = blockIdx.x * 256 + threadIdx.x;
  if (t >= nRays) return;
  int g = rayBase + t;
  const float* s = colbuf + (size_t)t * 9;   // [i][k]
  float* o = outcol + (size_t)g * 9;         // [o][k]
#pragma unroll
  for (int oo = 0; oo < 3; ++oo)
#pragma unroll
    for (int kk2 = 0; kk2 < 3; ++kk2)
      o[oo * 3 + kk2] = mix[oo * 3 + 0] * s[0 * 3 + kk2]
                      + mix[oo * 3 + 1] * s[1 * 3 + kk2]
                      + mix[oo * 3 + 2] * s[2 * 3 + kk2];
}

// ---------------------------------------------------------------------------
// Host launch
// ---------------------------------------------------------------------------
extern "C" void kernel_launch(void* const* d_in, const int* in_sizes, int n_in,
                              void* d_out, int out_size, void* d_ws, size_t ws_size,
                              hipStream_t stream)
{
  (void)n_in; (void)out_size; (void)ws_size;
  const int B = in_sizes[0];
  const int CHUNK = 4096;

  // inputs (jax pytree order: y,x,ray_origin,ray_dir,cam_feat,
  //         params{alpha,cB,cG,cR,delta,mix,pos,table} sorted keys)
  const float* ys  = (const float*)d_in[0];
  const float* xs  = (const float*)d_in[1];
  const float* ro  = (const float*)d_in[2];
  const float* rd  = (const float*)d_in[3];
  const float* cam = (const float*)d_in[4];
  const float* mix = (const float*)d_in[45];
  const float* table = (const float*)d_in[52];

  float* out_col   = (float*)d_out;                 // (B,3,3)
  float* out_alpha = out_col + (size_t)B * 9;       // (B,)
  float* out_delta = out_alpha + B;                 // (B,9)

  struct LW { int wi, bi, Kin, Kpad, N, Npad; };
  const LW L[23] = {
    {46,47, 64, 64, 512,512},   // 0  pos1 (sin)
    {48,49,512,512, 512,512},   // 1  pos2 (sin)
    {50,51,512,512, 384,384},   // 2  pos3 (sin) -> base
    {35,36,384,384, 256,256},   // 3  delta1 (relu)
    {37,38,256,256, 256,256},   // 4  delta2
    {39,40,256,256, 256,256},   // 5  delta3
    {41,42,256,256, 256,256},   // 6  delta4
    {43,44,256,256,   9, 16},   // 7  delta5 (tanh-clip)
    {27,28,486,512, 768,768},   // 8  cR1 (sin+relu)
    {29,30,768,768, 768,768},   // 9  cR2
    {31,32,768,768, 768,768},   // 10 cR3
    {33,34,768,768,   3, 16},   // 11 cR4 (sigmoid)
    {19,20,486,512, 768,768},   // 12 cG1
    {21,22,768,768, 768,768},   // 13
    {23,24,768,768, 768,768},   // 14
    {25,26,768,768,   3, 16},   // 15
    {11,12,486,512, 768,768},   // 16 cB1
    {13,14,768,768, 768,768},   // 17
    {15,16,768,768, 768,768},   // 18
    {17,18,768,768,   3, 16},   // 19
    { 5, 6,486,512, 256,256},   // 20 alpha1 (relu)
    { 7, 8,256,256, 256,256},   // 21 alpha2 (relu)
    { 9,10,256,256,   1, 16},   // 22 alpha3 (sigmoid)
  };

  size_t wOff[23], bOff[23], woff = 0, boff = 0;
  for (int i = 0; i < 23; ++i) {
    wOff[i] = woff; woff += (size_t)L[i].Kpad * L[i].Npad;
    bOff[i] = boff; boff += L[i].Npad;
  }

  char* wsb = (char*)d_ws;
  size_t off = 0;
  auto carve = [&](size_t bytes) -> void* {
    off = (off + 255) & ~(size_t)255;
    void* p = wsb + off; off += bytes; return p;
  };
  f16*   WH    = (f16*)carve(woff * 2);
  float* BS    = (float*)carve(boff * 4);
  f16*   h0    = (f16*)carve((size_t)CHUNK * 64  * 2);
  f16*   a1    = (f16*)carve((size_t)CHUNK * 512 * 2);
  f16*   a2    = (f16*)carve((size_t)CHUNK * 512 * 2);
  f16*   baseb = (f16*)carve((size_t)CHUNK * 384 * 2);
  f16*   d1    = (f16*)carve((size_t)CHUNK * 256 * 2);
  f16*   d2    = (f16*)carve((size_t)CHUNK * 256 * 2);
  f16*   inp   = (f16*)carve((size_t)CHUNK * 1536 * 2);
  f16*   c1    = (f16*)carve((size_t)CHUNK * 768 * 2);
  f16*   c2    = (f16*)carve((size_t)CHUNK * 768 * 2);
  float* colbuf = (float*)carve((size_t)CHUNK * 9 * 4);

  // pack weights/biases to padded f16 once per launch
  for (int i = 0; i < 23; ++i) {
    int tot = L[i].Kpad * L[i].Npad;
    pack_weight<<<(tot + 255) / 256, 256, 0, stream>>>(
        (const float*)d_in[L[i].wi], WH + wOff[i], L[i].Kin, L[i].Kpad, L[i].N, L[i].Npad);
    pack_bias<<<(L[i].Npad + 255) / 256, 256, 0, stream>>>(
        (const float*)d_in[L[i].bi], BS + bOff[i], L[i].N, L[i].Npad);
  }

  const int MT = CHUNK / 32;                 // 32-row tiles per chunk
  auto gy = [](int Npad) { return (Npad + 255) / 256; };
  auto sh = [](int Kpad) { return (size_t)32 * Kpad * 2; };

  for (int c0 = 0; c0 < B; c0 += CHUNK) {
    hash_concat<<<(CHUNK + 255) / 256, 256, 0, stream>>>(xs, ys, cam, table, h0, c0, CHUNK);

    gemm_wmma<ACT_SIN,true><<<dim3(MT,gy(512)),256,sh(64),stream>>>(h0,  64, WH+wOff[0], BS+bOff[0], a1,   512,512,  64,512);
    gemm_wmma<ACT_SIN,true><<<dim3(MT,gy(512)),256,sh(512),stream>>>(a1, 512, WH+wOff[1], BS+bOff[1], a2,   512,512, 512,512);
    gemm_wmma<ACT_SIN,true><<<dim3(MT,gy(384)),256,sh(512),stream>>>(a2, 512, WH+wOff[2], BS+bOff[2], baseb,384,384, 512,384);

    gemm_wmma<ACT_RELU,true><<<dim3(MT,gy(256)),256,sh(384),stream>>>(baseb,384, WH+wOff[3], BS+bOff[3], d1,256,256, 384,256);
    gemm_wmma<ACT_RELU,true><<<dim3(MT,gy(256)),256,sh(256),stream>>>(d1,  256, WH+wOff[4], BS+bOff[4], d2,256,256, 256,256);
    gemm_wmma<ACT_RELU,true><<<dim3(MT,gy(256)),256,sh(256),stream>>>(d2,  256, WH+wOff[5], BS+bOff[5], d1,256,256, 256,256);
    gemm_wmma<ACT_RELU,true><<<dim3(MT,gy(256)),256,sh(256),stream>>>(d1,  256, WH+wOff[6], BS+bOff[6], d2,256,256, 256,256);
    gemm_wmma<ACT_TANH_CLIP,false><<<dim3(MT,1),256,sh(256),stream>>>(d2,256, WH+wOff[7], BS+bOff[7],
                                                                      out_delta + (size_t)c0*9, 9, 9, 256,16);

    build_inp<<<(CHUNK*3 + 255) / 256, 256, 0, stream>>>(out_delta, rd, ro, baseb, inp, c0, CHUNK);

    const int CI[3] = {8, 12, 16};   // cR, cG, cB layer-table bases
    for (int ch = 0; ch < 3; ++ch) {
      int li = CI[ch];
      gemm_wmma<ACT_SIN_RELU,true><<<dim3(MT,gy(768)),256,sh(512),stream>>>(inp + ch*512, 1536,
          WH+wOff[li], BS+bOff[li], c1, 768,768, 512,768);
      gemm_wmma<ACT_RELU,true><<<dim3(MT,gy(768)),256,sh(768),stream>>>(c1, 768,
          WH+wOff[li+1], BS+bOff[li+1], c2, 768,768, 768,768);
      gemm_wmma<ACT_RELU,true><<<dim3(MT,gy(768)),256,sh(768),stream>>>(c2, 768,
          WH+wOff[li+2], BS+bOff[li+2], c1, 768,768, 768,768);
      gemm_wmma<ACT_SIGMOID,false><<<dim3(MT,1),256,sh(768),stream>>>(c1, 768,
          WH+wOff[li+3], BS+bOff[li+3], colbuf + ch*3, 9, 3, 768,16);
    }
    mix_sig<<<(CHUNK + 255) / 256, 256, 0, stream>>>(colbuf, mix, out_col, c0, CHUNK);

    gemm_wmma<ACT_RELU,true><<<dim3(MT,gy(256)),256,sh(512),stream>>>(inp, 1536, WH+wOff[20], BS+bOff[20], d1,256,256, 512,256);
    gemm_wmma<ACT_RELU,true><<<dim3(MT,gy(256)),256,sh(256),stream>>>(d1,  256, WH+wOff[21], BS+bOff[21], d2,256,256, 256,256);
    gemm_wmma<ACT_SIGMOID,false><<<dim3(MT,1),256,sh(256),stream>>>(d2, 256, WH+wOff[22], BS+bOff[22],
                                                                    out_alpha + c0, 1, 1, 256,16);
  }
}
